// LEGOTransformer_30030411333982
// MI455X (gfx1250) — matrix operand order repaired and checked by reading, not compiled, not run on hardware
//
#include <hip/hip_runtime.h>

// ---------------------------------------------------------------------------
// LEGO transformer forward with early-exit routing, MI455X (gfx1250, wave32).
// All matmuls use v_wmma_f32_16x16x32_bf16 with f32 accumulation.
// Double-buffered LDS pipeline; A-tile staged via global_load_async_to_lds
// (ASYNCcnt path, builtin confirmed present), B staged through VGPRs with
// f32->bf16 conversion.
// ---------------------------------------------------------------------------

typedef unsigned short u16;
typedef __attribute__((ext_vector_type(4)))  u16    u16x4;
typedef __attribute__((ext_vector_type(8)))  u16    u16x8;
typedef __attribute__((ext_vector_type(16))) u16    u16x16;
typedef __attribute__((ext_vector_type(16))) __bf16 bf16x16;
typedef __attribute__((ext_vector_type(8)))  float  f32x8;
typedef __attribute__((ext_vector_type(4)))  int    i32x4;

#define T_TOK  4096          // B*S = 4*1024 tokens
#define DIM    1024
#define DFF    4096
#define VOCAB  50257
#define NLAYER 4
#define TM     128
#define TN     128
#define TK     64
#define LDST   (TK + 8)      // padded LDS row stride (u16) to spread banks
#define ABUF   (TM * LDST)   // one A buffer, in u16 elements
#define BBUF   (TN * LDST)

#if defined(__has_builtin)
# if __has_builtin(__builtin_amdgcn_global_load_async_to_lds_b128)
#  define HAS_ASYNC_COPY 1
# endif
#endif
#ifndef HAS_ASYNC_COPY
# define HAS_ASYNC_COPY 0
#endif

#define GLOBAL_AS __attribute__((address_space(1)))
#define LDS_AS    __attribute__((address_space(3)))

__device__ __forceinline__ void wait_async0() {
#if defined(__has_builtin) && __has_builtin(__builtin_amdgcn_s_wait_asynccnt)
    __builtin_amdgcn_s_wait_asynccnt(0);
#else
    asm volatile("s_wait_asynccnt 0x0" ::: "memory");
#endif
}

#if HAS_ASYNC_COPY
__device__ __forceinline__ void async_copy_b128(const u16* gp, u16* lp) {
    // generic -> generic reinterpret, then same-pointee addrspace casts
    i32x4* g0 = (i32x4*)(void*)const_cast<u16*>(gp);
    i32x4* l0 = (i32x4*)(void*)lp;
    __builtin_amdgcn_global_load_async_to_lds_b128(
        (GLOBAL_AS i32x4*)g0, (LDS_AS i32x4*)l0, 0, 0);
}
#endif

__device__ __forceinline__ u16 f2bf(float f) {   // f32 -> bf16, round-nearest-even
    unsigned u = __float_as_uint(f);
    unsigned r = u + 0x7FFFu + ((u >> 16) & 1u);
    return (u16)(r >> 16);
}

__device__ __forceinline__ float gelu_tanh(float x) {
    const float c0 = 0.7978845608028654f;  // sqrt(2/pi)
    const float c1 = 0.044715f;
    float x3 = x * x * x;
    return 0.5f * x * (1.0f + tanhf(c0 * (x + c1 * x3)));
}

// Online softmax-stat merge: stats = { max (low 32), sumexp (high 32) }.
__device__ __forceinline__ void merge_stats(unsigned long long* p, float m2, float s2) {
    if (!(m2 > -__builtin_inff())) return;      // empty contribution
    unsigned long long old = *p, assumed;
    do {
        assumed  = old;
        float m1 = __uint_as_float((unsigned)(assumed & 0xffffffffull));
        float s1 = __uint_as_float((unsigned)(assumed >> 32));
        float m  = fmaxf(m1, m2);
        float e1 = (m1 > -__builtin_inff()) ? __expf(m1 - m) : 0.0f;
        float s  = s1 * e1 + s2 * __expf(m2 - m);
        unsigned long long nv =
            ((unsigned long long)__float_as_uint(s) << 32) |
            (unsigned long long)__float_as_uint(m);
        old = atomicCAS(p, assumed, nv);
    } while (old != assumed);
}

// ---------------------------------------------------------------------------
// Tiled bf16 WMMA GEMM: C[M,N] = A[M,K](bf16) * B (f32, converted in staging).
//   NMAJOR=false: B is [K,N] row-major (layer weights)
//   NMAJOR=true : B is [N,K] row-major (head_w, logits = h @ head_w^T)
// MODE: 0 = store bf16            (outBF)
//       1 = outF += acc (+bias); optional bf16 mirror (outBF2)
//       2 = store bf16(gelu(acc + bias))
//       3 = store f32 logits + online softmax stats merge
//       4 = store f32 logits only where mask[row] != 0
// ---------------------------------------------------------------------------
template <int MODE, bool NMAJOR>
__global__ __launch_bounds__(256)
void gemm_bf16_wmma(const u16* __restrict__ A, const float* __restrict__ B,
                    int M, int N, int K,
                    float* __restrict__ outF, u16* __restrict__ outBF,
                    const float* __restrict__ bias, u16* __restrict__ outBF2,
                    unsigned long long* __restrict__ stats,
                    const int* __restrict__ mask) {
    __shared__ u16 ldsA[2 * ABUF];
    __shared__ u16 ldsB[2 * BBUF];   // stored n-major: ldsB[n][k]

    const int tid   = threadIdx.x;
    const int lane  = tid & 31;
    const int wid   = tid >> 5;
    const int wm    = wid >> 2;       // 0..1  : wave row block (64 rows)
    const int wn    = wid & 3;        // 0..3  : wave col block (32 cols)
    const int tileM = blockIdx.y * TM;
    const int tileN = blockIdx.x * TN;

    f32x8 acc[4][2];
    const f32x8 zero = {0.f, 0.f, 0.f, 0.f, 0.f, 0.f, 0.f, 0.f};
#pragma unroll
    for (int i = 0; i < 4; ++i)
#pragma unroll
        for (int j = 0; j < 2; ++j) acc[i][j] = zero;

    u16x8  ra[4];    // A staging registers (fallback path only)
    float4 rb[8];    // B staging registers (f32, converted at store time)

    // ---- A tile staging: async DMA to LDS if available, else reg-staged ----
    auto issueA = [&](int kt, int buf) {
#pragma unroll
        for (int it = 0; it < 4; ++it) {
            int idx = it * 256 + tid;            // 0..1023
            int r   = idx >> 3;                  // 0..127
            int c8  = (idx & 7) * 8;             // 0..56
            const u16* gp = A + (size_t)(tileM + r) * K + kt + c8;
#if HAS_ASYNC_COPY
            async_copy_b128(gp, &ldsA[buf * ABUF + r * LDST + c8]);
#else
            ra[it] = *(const u16x8*)gp;
#endif
        }
        (void)buf;
    };
    auto storeA = [&](int buf) {
#if !HAS_ASYNC_COPY
#pragma unroll
        for (int it = 0; it < 4; ++it) {
            int idx = it * 256 + tid;
            int r   = idx >> 3;
            int c8  = (idx & 7) * 8;
            *(u16x8*)(&ldsA[buf * ABUF + r * LDST + c8]) = ra[it];
        }
#endif
        (void)buf;
    };

    // ---- B tile staging: batched reg loads, convert f32->bf16 on store ----
    auto loadB = [&](int kt) {
        if (NMAJOR) {
#pragma unroll
            for (int it = 0; it < 8; ++it) {
                int idx = it * 256 + tid;        // 0..2047
                int n   = idx >> 4;              // 0..127
                int k4  = (idx & 15) * 4;        // 0..60
                int gN  = tileN + n;
                rb[it] = (gN < N) ? *(const float4*)(B + (size_t)gN * K + kt + k4)
                                  : make_float4(0.f, 0.f, 0.f, 0.f);
            }
        } else {
#pragma unroll
            for (int it = 0; it < 8; ++it) {
                int idx = it * 256 + tid;        // 0..2047
                int k   = idx >> 5;              // 0..63
                int n4  = (idx & 31) * 4;        // 0..124
                rb[it] = *(const float4*)(B + (size_t)(kt + k) * N + tileN + n4);
            }
        }
        // prefetch the tile after next (emits global_prefetch_b8)
        if (kt + TK < K) {
            if (NMAJOR)
                __builtin_prefetch(B + (size_t)(tileN + (tid >> 1)) * K + kt + TK, 0, 3);
            else
                __builtin_prefetch(B + (size_t)(kt + TK + (tid >> 2)) * N + tileN, 0, 3);
        }
    };
    auto storeB = [&](int buf) {
        if (NMAJOR) {
#pragma unroll
            for (int it = 0; it < 8; ++it) {
                int idx = it * 256 + tid;
                int n   = idx >> 4;
                int k4  = (idx & 15) * 4;
                u16x4 o;
                o[0] = f2bf(rb[it].x); o[1] = f2bf(rb[it].y);
                o[2] = f2bf(rb[it].z); o[3] = f2bf(rb[it].w);
                *(u16x4*)(&ldsB[buf * BBUF + n * LDST + k4]) = o;
            }
        } else {
#pragma unroll
            for (int it = 0; it < 8; ++it) {
                int idx = it * 256 + tid;
                int k   = idx >> 5;
                int n4  = (idx & 31) * 4;
                u16* base = &ldsB[buf * BBUF];
                base[(n4 + 0) * LDST + k] = f2bf(rb[it].x);
                base[(n4 + 1) * LDST + k] = f2bf(rb[it].y);
                base[(n4 + 2) * LDST + k] = f2bf(rb[it].z);
                base[(n4 + 3) * LDST + k] = f2bf(rb[it].w);
            }
        }
    };

    // ---- 16 WMMAs on one staged K-tile ----
    auto computeTile = [&](int buf) {
        const u16* bA = &ldsA[buf * ABUF];
        const u16* bB = &ldsB[buf * BBUF];
#pragma unroll
        for (int kk = 0; kk < TK; kk += 32) {
            bf16x16 aF[4], bF[2];
            const int am    = lane & 15;
            const int khalf = (lane >> 4) * 8;   // A: lanes 0-15 hold K 0-7/16-23
#pragma unroll
            for (int fm = 0; fm < 4; ++fm) {
                const u16* p = &bA[(wm * 64 + fm * 16 + am) * LDST + kk + khalf];
                u16x8 lo = *(const u16x8*)p;
                u16x8 hi = *(const u16x8*)(p + 16);
                u16x16 t;
#pragma unroll
                for (int e = 0; e < 8; ++e) { t[e] = lo[e]; t[8 + e] = hi[e]; }
                aF[fm] = __builtin_bit_cast(bf16x16, t);
            }
            const int klo = (lane >> 4) * 8;     // B: lanes 0-15 K 0-7/16-23, 16-31 K 8-15/24-31
#pragma unroll
            for (int fn = 0; fn < 2; ++fn) {
                const u16* p = &bB[(wn * 32 + fn * 16 + am) * LDST + kk + klo];
                u16x8 lo = *(const u16x8*)p;
                u16x8 hi = *(const u16x8*)(p + 16);
                u16x16 t;
#pragma unroll
                for (int e = 0; e < 8; ++e) { t[e] = lo[e]; t[8 + e] = hi[e]; }
                bF[fn] = __builtin_bit_cast(bf16x16, t);
            }
#pragma unroll
            for (int fm = 0; fm < 4; ++fm)
#pragma unroll
                for (int fn = 0; fn < 2; ++fn)
                    acc[fm][fn] = __builtin_amdgcn_wmma_f32_16x16x32_bf16(
                        false, aF[fm], false, bF[fn], (short)0, acc[fm][fn],
                        false, false);
        }
    };

    // ---- software pipeline: one barrier per K-tile -------------------------
    const int nt = K / TK;
    issueA(0, 0);
    loadB(0);
    storeA(0);
    storeB(0);
#if HAS_ASYNC_COPY
    wait_async0();
#endif
    int cb = 0;
    for (int i = 0; i < nt; ++i) {
        __syncthreads();                          // buf cb ready & LDS quiesced
        const bool hasNext = (i + 1) < nt;
        if (hasNext) {
            issueA((i + 1) * TK, cb ^ 1);         // async writes overlap compute
            loadB((i + 1) * TK);                  // global loads overlap compute
        }
        computeTile(cb);
        if (hasNext) {
            storeA(cb ^ 1);                       // (fallback path only)
            storeB(cb ^ 1);                       // forces one batched loadcnt wait
#if HAS_ASYNC_COPY
            wait_async0();
#endif
        }
        cb ^= 1;
    }

    // ------------------------------ epilogue -------------------------------
    const int nhalf = lane & 15;          // C: lanes 0-15 N=lane, VGPR j -> M=j
    const int mjoff = (lane >> 4) * 8;    //    lanes 16-31: M=j+8

    if (MODE == 3) {
        // store all logits + merge per-row softmax stats (max, sumexp)
#pragma unroll
        for (int fm = 0; fm < 4; ++fm) {
#pragma unroll
            for (int j = 0; j < 8; ++j) {
                int   mRow = tileM + wm * 64 + fm * 16 + mjoff + j;
                int   n0   = tileN + wn * 32 + nhalf;
                int   n1   = n0 + 16;
                float v0   = acc[fm][0][j];
                float v1   = acc[fm][1][j];
                if (n0 < N) outF[(size_t)mRow * N + n0] = v0;
                if (n1 < N) outF[(size_t)mRow * N + n1] = v1;
                float lm = fmaxf(n0 < N ? v0 : -__builtin_inff(),
                                 n1 < N ? v1 : -__builtin_inff());
#pragma unroll
                for (int off = 1; off < 16; off <<= 1)
                    lm = fmaxf(lm, __shfl_xor(lm, off, 32));
                float ls = (n0 < N ? __expf(v0 - lm) : 0.f) +
                           (n1 < N ? __expf(v1 - lm) : 0.f);
#pragma unroll
                for (int off = 1; off < 16; off <<= 1)
                    ls += __shfl_xor(ls, off, 32);
                if (nhalf == 0) merge_stats(stats + mRow, lm, ls);
            }
        }
        return;
    }

    const bool wbf2 = (MODE == 1) && (outBF2 != nullptr);
#pragma unroll
    for (int fm = 0; fm < 4; ++fm) {
#pragma unroll
        for (int fn = 0; fn < 2; ++fn) {
            int   nCol = tileN + wn * 32 + fn * 16 + nhalf;
            float bv   = 0.f;
            if ((MODE == 1 || MODE == 2) && bias != nullptr && nCol < N)
                bv = bias[nCol];
#pragma unroll
            for (int j = 0; j < 8; ++j) {
                int   mRow = tileM + wm * 64 + fm * 16 + mjoff + j;
                float v    = acc[fm][fn][j];
                if (MODE == 0) {
                    outBF[(size_t)mRow * N + nCol] = f2bf(v);
                } else if (MODE == 1) {
                    size_t o  = (size_t)mRow * N + nCol;
                    float  nv = outF[o] + v + bv;
                    outF[o] = nv;
                    if (wbf2) outBF2[o] = f2bf(nv);
                } else if (MODE == 2) {
                    outBF[(size_t)mRow * N + nCol] = f2bf(gelu_tanh(v + bv));
                } else if (MODE == 4) {
                    if (nCol < N && mask[mRow]) outF[(size_t)mRow * N + nCol] = v;
                }
            }
        }
    }
}

// ---------------------------------------------------------------------------
// helper kernels
// ---------------------------------------------------------------------------
__global__ __launch_bounds__(256)
void k_gather(const int* __restrict__ x, const float* __restrict__ emb,
              float* __restrict__ h) {
    int t   = blockIdx.x;
    int row = x[t];
    ((float4*)(h + (size_t)t * DIM))[threadIdx.x] =
        ((const float4*)(emb + (size_t)row * DIM))[threadIdx.x];
}

__global__ __launch_bounds__(256)
void k_ln(const float* __restrict__ h, const float* __restrict__ sc,
          const float* __restrict__ bi, u16* __restrict__ out) {
    __shared__ float rsum[256], rsq[256];
    const int t   = blockIdx.x;
    const int tid = threadIdx.x;
    float4 x = ((const float4*)(h + (size_t)t * DIM))[tid];
    rsum[tid] = x.x + x.y + x.z + x.w;
    rsq[tid]  = x.x * x.x + x.y * x.y + x.z * x.z + x.w * x.w;
    __syncthreads();
    for (int o = 128; o > 0; o >>= 1) {
        if (tid < o) { rsum[tid] += rsum[tid + o]; rsq[tid] += rsq[tid + o]; }
        __syncthreads();
    }
    float mean = rsum[0] * (1.0f / DIM);
    float var  = rsq[0] * (1.0f / DIM) - mean * mean;
    float r    = rsqrtf(var + 1e-5f);
    float4 scv = ((const float4*)sc)[tid];
    float4 biv = ((const float4*)bi)[tid];
    u16x4 o4;
    o4[0] = f2bf((x.x - mean) * r * scv.x + biv.x);
    o4[1] = f2bf((x.y - mean) * r * scv.y + biv.y);
    o4[2] = f2bf((x.z - mean) * r * scv.z + biv.z);
    o4[3] = f2bf((x.w - mean) * r * scv.w + biv.w);
    *(u16x4*)(out + (size_t)t * DIM + tid * 4) = o4;
}

__global__ void k_stats_init(unsigned long long* __restrict__ st) {
    int t = blockIdx.x * 256 + threadIdx.x;
    if (t < T_TOK) st[t] = 0x00000000FF800000ull;   // { max=-inf, sumexp=0 }
}

__global__ void k_route(const unsigned long long* __restrict__ st,
                        int* __restrict__ keep, float th) {
    int t = blockIdx.x * 256 + threadIdx.x;
    if (t < T_TOK) {
        float s  = __uint_as_float((unsigned)(st[t] >> 32));
        float mp = 1.0f / s;                 // max softmax prob = 1 / sum(exp(l-max))
        keep[t]  = (mp >= th) ? 0 : 1;       // 1 => still active after block 0
    }
}

// ---------------------------------------------------------------------------
// orchestration
// ---------------------------------------------------------------------------
extern "C" void kernel_launch(void* const* d_in, const int* in_sizes, int n_in,
                              void* d_out, int out_size, void* d_ws, size_t ws_size,
                              hipStream_t stream) {
    (void)in_sizes; (void)n_in; (void)out_size; (void)ws_size;
    const int*   x      = (const int*)  d_in[0];
    const float* emb    = (const float*)d_in[1];
    const float* head_w = (const float*)d_in[2];
    const float* ln1_s  = (const float*)d_in[3];
    const float* ln1_b  = (const float*)d_in[4];
    const float* wv     = (const float*)d_in[5];
    const float* wo     = (const float*)d_in[6];
    const float* ln2_s  = (const float*)d_in[7];
    const float* ln2_b  = (const float*)d_in[8];
    const float* w1     = (const float*)d_in[9];
    const float* b1     = (const float*)d_in[10];
    const float* w2     = (const float*)d_in[11];
    const float* b2     = (const float*)d_in[12];
    float* out = (float*)d_out;

    char* p = (char*)d_ws;
    auto carve = [&](size_t bytes) {
        char* r = p;
        p += (bytes + 255) & ~(size_t)255;
        return r;
    };
    float* h    = (float*)carve((size_t)T_TOK * DIM * 4);   // residual stream (f32)
    u16*   h_bf = (u16*)  carve((size_t)T_TOK * DIM * 2);   // bf16 mirror for head
    u16*   a_bf = (u16*)  carve((size_t)T_TOK * DIM * 2);   // LN output
    u16*   t_bf = (u16*)  carve((size_t)T_TOK * DIM * 2);   // a @ wv
    u16*   g_bf = (u16*)  carve((size_t)T_TOK * DFF * 2);   // gelu(m @ w1 + b1)
    unsigned long long* stats = (unsigned long long*)carve((size_t)T_TOK * 8);
    int*   keep = (int*)  carve((size_t)T_TOK * 4);

    dim3 blk(256);
    k_gather<<<T_TOK, blk, 0, stream>>>(x, emb, h);

    const int vtiles = (VOCAB + TN - 1) / TN;
    for (int li = 0; li < NLAYER; ++li) {
        const bool blockEnd = (li == 1) || (li == 3);
        // attention (seq-len-1): h += LN1(h) @ wv @ wo
        k_ln<<<T_TOK, blk, 0, stream>>>(h, ln1_s + li * DIM, ln1_b + li * DIM, a_bf);
        gemm_bf16_wmma<0, false><<<dim3(DIM / TN, T_TOK / TM), blk, 0, stream>>>(
            a_bf, wv + (size_t)li * DIM * DIM, T_TOK, DIM, DIM,
            nullptr, t_bf, nullptr, nullptr, nullptr, nullptr);
        gemm_bf16_wmma<1, false><<<dim3(DIM / TN, T_TOK / TM), blk, 0, stream>>>(
            t_bf, wo + (size_t)li * DIM * DIM, T_TOK, DIM, DIM,
            h, nullptr, nullptr, nullptr, nullptr, nullptr);
        // MLP: h += gelu(LN2(h) @ w1 + b1) @ w2 + b2
        k_ln<<<T_TOK, blk, 0, stream>>>(h, ln2_s + li * DIM, ln2_b + li * DIM, a_bf);
        gemm_bf16_wmma<2, false><<<dim3(DFF / TN, T_TOK / TM), blk, 0, stream>>>(
            a_bf, w1 + (size_t)li * DIM * DFF, T_TOK, DFF, DIM,
            nullptr, g_bf, b1 + (size_t)li * DFF, nullptr, nullptr, nullptr);
        gemm_bf16_wmma<1, false><<<dim3(DIM / TN, T_TOK / TM), blk, 0, stream>>>(
            g_bf, w2 + (size_t)li * DFF * DIM, T_TOK, DIM, DFF,
            h, nullptr, b2 + (size_t)li * DIM,
            blockEnd ? h_bf : nullptr, nullptr, nullptr);

        if (li == 1) {
            // head pass 0: write ALL logits + softmax stats, then routing
            k_stats_init<<<(T_TOK + 255) / 256, blk, 0, stream>>>(stats);
            gemm_bf16_wmma<3, true><<<dim3(vtiles, T_TOK / TM), blk, 0, stream>>>(
                h_bf, head_w, T_TOK, VOCAB, DIM,
                out, nullptr, nullptr, nullptr, stats, nullptr);
            k_route<<<(T_TOK + 255) / 256, blk, 0, stream>>>(stats, keep, 1e-4f);
        } else if (li == 3) {
            // head pass 1: overwrite logits only for tokens that did not exit
            gemm_bf16_wmma<4, true><<<dim3(vtiles, T_TOK / TM), blk, 0, stream>>>(
                h_bf, head_w, T_TOK, VOCAB, DIM,
                out, nullptr, nullptr, nullptr, nullptr, keep);
        }
    }
}
